// InfoNCE_46918222741935
// MI455X (gfx1250) — compile-verified, matrix operand order checked by reading, not run on hardware
//
#include <hip/hip_runtime.h>
#include <hip/hip_bf16.h>
#include <math.h>

typedef __attribute__((ext_vector_type(16))) _Float16 v16h;
typedef __attribute__((ext_vector_type(8)))  float    v8f;

#define D_DIM   256
#define KT      8                       // 256 / 32 k-steps
#define WAVES   8
#define BROWS   128                     // WAVES * 16 anchor rows per block
#define NSPLIT  8                       // column splits for occupancy
#define INV_T   14.285714285714286f
#define K2      20.60992915555662f      // (1/T) * log2(e)
#define LN2     0.6931471805599453f

#define TILE_NROWS 32                   // pn rows staged per iteration (2 subtiles)
#define ROW_BYTES  (D_DIM * 2)          // 512
#define ROW_STRIDE (ROW_BYTES + 16)     // 528: pad to kill LDS bank conflicts
#define BUF_BYTES  (TILE_NROWS * ROW_STRIDE)   // 16896
#define BUF_U4     (BUF_BYTES / 16)

union FragAB {
    v16h  h;
    uint4 u[2];
};

// ---------------- K0: L2 normalize rows of f32 [nrows, 256] -> f16 ----------------
__global__ void norm_kernel(const float* __restrict__ X, _Float16* __restrict__ Y, int nrows) {
    const int row  = blockIdx.x * 8 + (threadIdx.x >> 5);
    const int lane = threadIdx.x & 31;
    if (row >= nrows) return;
    const float4* px = (const float4*)(X + (size_t)row * D_DIM) + lane * 2;
    float4 x0 = px[0];
    float4 x1 = px[1];
    float ss = x0.x*x0.x + x0.y*x0.y + x0.z*x0.z + x0.w*x0.w
             + x1.x*x1.x + x1.y*x1.y + x1.z*x1.z + x1.w*x1.w;
#pragma unroll
    for (int off = 16; off > 0; off >>= 1) ss += __shfl_xor(ss, off, 32);
    float inv = 1.0f / fmaxf(sqrtf(ss), 1e-12f);
    union { _Float16 h[8]; uint4 u; } o;
    o.h[0] = (_Float16)(x0.x * inv); o.h[1] = (_Float16)(x0.y * inv);
    o.h[2] = (_Float16)(x0.z * inv); o.h[3] = (_Float16)(x0.w * inv);
    o.h[4] = (_Float16)(x1.x * inv); o.h[5] = (_Float16)(x1.y * inv);
    o.h[6] = (_Float16)(x1.z * inv); o.h[7] = (_Float16)(x1.w * inv);
    ((uint4*)(Y + (size_t)row * D_DIM))[lane] = o.u;
}

__global__ void init_acc_kernel(float* acc) { acc[0] = 0.0f; acc[1] = 0.0f; }
__global__ void final_kernel(const float* acc, float* out) { out[0] = acc[0] / acc[1]; }

// A fragment from global: lane holds 16 halves at K offsets {8h..8h+7, 8h+16..8h+23}
__device__ __forceinline__ FragAB load_frag_g(const _Float16* __restrict__ base, int kt, int h) {
    FragAB f;
    const uint4* p = (const uint4*)base;   // 8 halves per uint4
    f.u[0] = p[kt * 4 + h];
    f.u[1] = p[kt * 4 + h + 2];
    return f;
}

// B fragment from padded LDS tile: row = sub*16 + l16, stride 528 B
__device__ __forceinline__ FragAB load_frag_s(const char* smBase, int lrow, int kt, int h) {
    FragAB f;
    const uint4* p = (const uint4*)(smBase + lrow * ROW_STRIDE);
    f.u[0] = p[kt * 4 + h];
    f.u[1] = p[kt * 4 + h + 2];
    return f;
}

// Cooperative async stage of one 32-row B tile into a padded LDS buffer.
// 1024 b128 copies; each of 256 threads issues 4 ASYNCcnt-tracked copies.
__device__ __forceinline__ void issue_tile_async(const _Float16* __restrict__ Ph, int jt,
                                                 unsigned ldsBase, int tid) {
    const char* gbase = (const char*)(Ph + (size_t)jt * TILE_NROWS * D_DIM);
#pragma unroll
    for (int k = 0; k < 4; ++k) {
        int idx = tid + k * 256;                        // 0..1023
        int row = idx >> 5, col = idx & 31;             // 32 x uint4 per row
        unsigned l = ldsBase + (unsigned)(row * ROW_STRIDE + col * 16);
        const char* g = gbase + (size_t)idx * 16;
        asm volatile("global_load_async_to_lds_b128 %0, %1, off"
                     :: "v"(l), "v"(g) : "memory");
    }
}

__device__ __forceinline__ void wait_async0() {
    asm volatile("s_wait_asynccnt 0x0" ::: "memory");
}

// ---------------- K1: per-row fixed-shift sum-exp over negatives ----------------
// lse_neg(row) = INV_T + log( sum_j !match exp(S - INV_T) );  |S| <= INV_T so no over/underflow.
__global__ void stats_kernel(const _Float16* __restrict__ Ah, const _Float16* __restrict__ Ph,
                             const int* __restrict__ labA, const int* __restrict__ labP,
                             float* __restrict__ rowsum, int B, int M) {
    __shared__ uint4 smem4[2 * BUF_U4];                 // 2 x 16.5 KB padded buffers
    const int tid  = threadIdx.x;
    const int lane = tid & 31;
    const int wave = tid >> 5;
    const int h    = lane >> 4;
    const int l16  = lane & 15;

    const int tileRow  = blockIdx.x * BROWS + wave * 16;
    const int arow     = tileRow + l16;
    const int crowBase = tileRow + 8 * h;

    FragAB a[KT];
    const _Float16* abase = Ah + (size_t)arow * D_DIM;
#pragma unroll
    for (int kt = 0; kt < KT; ++kt) a[kt] = load_frag_g(abase, kt, h);

    int la[8];
#pragma unroll
    for (int r = 0; r < 8; ++r) la[r] = labA[crowBase + r];

    float s[8];
#pragma unroll
    for (int r = 0; r < 8; ++r) s[r] = 0.0f;

    const int tiles = (M / TILE_NROWS) / NSPLIT;
    const int jt0 = blockIdx.y * tiles;
    const int jt1 = jt0 + tiles;
    const unsigned ldsBase = (unsigned)(uintptr_t)(&smem4[0]);

    issue_tile_async(Ph, jt0, ldsBase, tid);
    for (int jt = jt0; jt < jt1; ++jt) {
        const int cur = (jt - jt0) & 1;
        wait_async0();
        __syncthreads();
        if (jt + 1 < jt1)
            issue_tile_async(Ph, jt + 1, ldsBase + (cur ^ 1) * BUF_BYTES, tid);

        const char* smBase = (const char*)smem4 + cur * BUF_BYTES;
#pragma unroll
        for (int sub = 0; sub < 2; ++sub) {
            const int lp = labP[jt * TILE_NROWS + sub * 16 + l16];
            FragAB b[KT];
#pragma unroll
            for (int kt = 0; kt < KT; ++kt) b[kt] = load_frag_s(smBase, sub * 16 + l16, kt, h);
            v8f c = {};
#pragma unroll
            for (int kt = 0; kt < KT; ++kt)
                c = __builtin_amdgcn_wmma_f32_16x16x32_f16(false, a[kt].h, false, b[kt].h,
                                                           (short)0, c, false, false);
#pragma unroll
            for (int r = 0; r < 8; ++r) {
                // exp(S - 1/T) == 2^(c*K2 - K2), one fma + one v_exp_f32
                float e = __builtin_amdgcn_exp2f(__builtin_fmaf(c[r], K2, -K2));
                s[r] += (la[r] == lp) ? 0.0f : e;
            }
        }
    }
#pragma unroll
    for (int off = 1; off < 16; off <<= 1) {
#pragma unroll
        for (int r = 0; r < 8; ++r) s[r] += __shfl_xor(s[r], off, 32);
    }
    if (l16 == 0) {
#pragma unroll
        for (int r = 0; r < 8; ++r)
            rowsum[(size_t)blockIdx.y * B + crowBase + r] = s[r];
    }
}

// ---------------- K2: recompute S only at subtiles containing positives ----------------
__global__ void loss_kernel(const _Float16* __restrict__ Ah, const _Float16* __restrict__ Ph,
                            const int* __restrict__ labA, const int* __restrict__ labP,
                            const float* __restrict__ rowsum, float* __restrict__ acc,
                            int B, int M) {
    __shared__ uint4 smem4[2 * BUF_U4];
    const int tid  = threadIdx.x;
    const int lane = tid & 31;
    const int wave = tid >> 5;
    const int h    = lane >> 4;
    const int l16  = lane & 15;

    const int tileRow  = blockIdx.x * BROWS + wave * 16;
    const int arow     = tileRow + l16;
    const int crowBase = tileRow + 8 * h;

    FragAB a[KT];
    const _Float16* abase = Ah + (size_t)arow * D_DIM;
#pragma unroll
    for (int kt = 0; kt < KT; ++kt) a[kt] = load_frag_g(abase, kt, h);

    int la[8];
    float lg2[8];                          // log2(sum_exp) per row
#pragma unroll
    for (int r = 0; r < 8; ++r) {
        const int row = crowBase + r;
        la[r] = labA[row];
        float ss = 0.0f;
        for (int sp = 0; sp < NSPLIT; ++sp) ss += rowsum[(size_t)sp * B + row];
        lg2[r] = __builtin_amdgcn_logf(ss);  // v_log_f32 (base 2); -inf iff no negatives
    }

    const int tiles = (M / TILE_NROWS) / NSPLIT;
    const int jt0 = blockIdx.y * tiles;
    const int jt1 = jt0 + tiles;
    const unsigned ldsBase = (unsigned)(uintptr_t)(&smem4[0]);

    float lossAcc = 0.0f, cnt = 0.0f;
    issue_tile_async(Ph, jt0, ldsBase, tid);
    for (int jt = jt0; jt < jt1; ++jt) {
        const int cur = (jt - jt0) & 1;
        wait_async0();
        __syncthreads();
        if (jt + 1 < jt1)
            issue_tile_async(Ph, jt + 1, ldsBase + (cur ^ 1) * BUF_BYTES, tid);

        const char* smBase = (const char*)smem4 + cur * BUF_BYTES;
#pragma unroll
        for (int sub = 0; sub < 2; ++sub) {
            const int lp = labP[jt * TILE_NROWS + sub * 16 + l16];
            bool anyr = false;
#pragma unroll
            for (int r = 0; r < 8; ++r) anyr |= (la[r] == lp);
            if (__any(anyr)) {             // ~22% of 16x16 subtiles contain a positive
                FragAB b[KT];
#pragma unroll
                for (int kt = 0; kt < KT; ++kt) b[kt] = load_frag_s(smBase, sub * 16 + l16, kt, h);
                v8f c = {};
#pragma unroll
                for (int kt = 0; kt < KT; ++kt)
                    c = __builtin_amdgcn_wmma_f32_16x16x32_f16(false, a[kt].h, false, b[kt].h,
                                                               (short)0, c, false, false);
#pragma unroll
                for (int r = 0; r < 8; ++r) {
                    if (la[r] == lp) {
                        // loss = ln(1 + 2^(lg2 - (c*K2 - K2))), all base-2 hardware ops
                        float d2 = lg2[r] - __builtin_fmaf(c[r], K2, -K2);
                        float p  = __builtin_amdgcn_exp2f(d2);
                        lossAcc += LN2 * __builtin_amdgcn_logf(1.0f + p);
                        cnt     += 1.0f;
                    }
                }
            }
        }
    }
#pragma unroll
    for (int off = 16; off > 0; off >>= 1) {
        lossAcc += __shfl_xor(lossAcc, off, 32);
        cnt     += __shfl_xor(cnt, off, 32);
    }
    if (lane == 0) {
        atomicAdd(&acc[0], lossAcc);
        atomicAdd(&acc[1], cnt);
    }
}

extern "C" void kernel_launch(void* const* d_in, const int* in_sizes, int n_in,
                              void* d_out, int out_size, void* d_ws, size_t ws_size,
                              hipStream_t stream) {
    const float* embA = (const float*)d_in[0];
    const float* embP = (const float*)d_in[1];
    const int*   labA = (const int*)d_in[2];
    const int*   labP = (const int*)d_in[3];
    const int B = in_sizes[0] / D_DIM;   // 8192
    const int M = in_sizes[1] / D_DIM;   // 8192

    char* w = (char*)d_ws;
    _Float16* Ah = (_Float16*)w;
    size_t offs = (size_t)B * D_DIM * sizeof(_Float16);
    _Float16* Ph = (_Float16*)(w + offs);
    offs += (size_t)M * D_DIM * sizeof(_Float16);
    float* rowsum = (float*)(w + offs);
    offs += (size_t)NSPLIT * B * sizeof(float);
    float* acc = (float*)(w + offs);

    init_acc_kernel<<<1, 1, 0, stream>>>(acc);
    norm_kernel<<<B / 8, 256, 0, stream>>>(embA, Ah, B);
    norm_kernel<<<M / 8, 256, 0, stream>>>(embP, Ph, M);

    dim3 grid(B / BROWS, NSPLIT);
    stats_kernel<<<grid, 256, 0, stream>>>(Ah, Ph, labA, labP, rowsum, B, M);
    loss_kernel<<<grid, 256, 0, stream>>>(Ah, Ph, labA, labP, rowsum, acc, B, M);
    final_kernel<<<1, 1, 0, stream>>>(acc, (float*)d_out);
}
